// tDBN_bv_1_55044300866042
// MI455X (gfx1250) — matrix operand order repaired and checked by reading, not compiled, not run on hardware
//
#include <hip/hip_runtime.h>
#include <hip/hip_bf16.h>

// ---------------------------------------------------------------------------
// MI455X (gfx1250) sparse-voxel 3D CNN backbone.
//   * Convs = implicit GEMM on v_wmma_f32_16x16x32_bf16 (fp32 accum).
//   * Weights pre-packed (bf16, zero-padded in taps and M-tiles) into
//     per-lane WMMA A-fragment order: A fragment = one 32B global load.
//   * B fragments from an LDS bf16 tile (BN+ReLU+mask fused at staging,
//     raw fp32 tile staged via gfx1250 async-to-LDS): one 32B LDS load.
//   * conv3 kernel templated on MTILES: MT>=4 -> waves own M-tiles;
//     MT<=2 -> split-K (taps) across waves + LDS reduction. All WMMA
//     control flow is compile-time uniform: no EXEC-masked WMMA regions.
// ---------------------------------------------------------------------------

typedef __attribute__((ext_vector_type(16))) __bf16 v16bf;
typedef __attribute__((ext_vector_type(8)))  float  v8f;

#define WT    16            // output W positions per block (N tile)
#define CBK   32            // channels per K-block (bf16 WMMA K=32)
#define CBP   33            // fp32 staging pitch (bank-conflict pad)
#define BP    40            // bf16 tile pitch (16B-aligned rows, padded)
#define ROWS3 162           // 3*3*(WT+2) patch rows for 3x3x3 s1 conv
#define NTAP3 28            // 27 taps + 1 zero-weight pad tap (guard-free split-K)

#if defined(__gfx1250__) && __has_builtin(__builtin_amdgcn_global_load_async_to_lds_b32) && __has_builtin(__builtin_amdgcn_s_wait_asynccnt)
#define HAVE_ASYNC 1
#else
#define HAVE_ASYNC 0
#endif

#if HAVE_ASYNC
typedef __attribute__((address_space(1))) int* gptr_i32;
typedef __attribute__((address_space(3))) int* lptr_i32;
#endif

// ------------------------------ utility kernels ----------------------------

__global__ void zero_kernel(float* p, long n) {
  long i = (long)blockIdx.x * blockDim.x + threadIdx.x;
  if (i < n) p[i] = 0.0f;
}

__global__ void scatter_kernel(const float* __restrict__ vf,
                               const int* __restrict__ coors,
                               float* __restrict__ x, float* __restrict__ mask,
                               int n, int D, int H, int W) {
  int i = blockIdx.x * blockDim.x + threadIdx.x;
  if (i >= n) return;
  int d = coors[4 * i + 1], h = coors[4 * i + 2], w = coors[4 * i + 3];
  long off = ((long)d * H + h) * W + w;
  atomicAdd(&x[off], vf[i]);
  mask[off] = 1.0f;
}

__global__ __launch_bounds__(256)
void stats_kernel(const float* __restrict__ x, const float* __restrict__ mask,
                  long nvox, int C, float* __restrict__ stats) {
  int c = blockIdx.y;
  __shared__ float ss[256], sq[256], sm[256];
  int t = threadIdx.x;
  float s = 0.f, q = 0.f, mc = 0.f;
  for (long i = (long)blockIdx.x * 256 + t; i < nvox; i += (long)gridDim.x * 256) {
    float m = mask[i];
    float v = x[(long)c * nvox + i];
    s += v * m; q += v * v * m; mc += m;
  }
  ss[t] = s; sq[t] = q; sm[t] = mc;
  __syncthreads();
  for (int off = 128; off > 0; off >>= 1) {
    if (t < off) { ss[t] += ss[t + off]; sq[t] += sq[t + off]; sm[t] += sm[t + off]; }
    __syncthreads();
  }
  if (t == 0) {
    atomicAdd(&stats[1 + c], ss[0]);
    atomicAdd(&stats[1 + C + c], sq[0]);
    if (c == 0) atomicAdd(&stats[0], sm[0]);
  }
}

__global__ void finalize_bn_kernel(const float* __restrict__ stats,
                                   const float* __restrict__ g,
                                   const float* __restrict__ b,
                                   float* __restrict__ scale,
                                   float* __restrict__ bias, int C) {
  int c = blockIdx.x * blockDim.x + threadIdx.x;
  if (c >= C) return;
  float cnt  = fmaxf(stats[0], 1.0f);
  float mean = stats[1 + c] / cnt;
  float var  = stats[1 + C + c] / cnt - mean * mean;
  float sc   = g[c] * rsqrtf(var + 1e-4f);
  scale[c] = sc;
  bias[c]  = b[c] - mean * sc;
}

__global__ void pool_mask_kernel(const float* __restrict__ in, float* __restrict__ out,
                                 int D, int H, int W, int Do, int Ho, int Wo,
                                 int wd, int wh, int ww, int sd, int sh, int sw) {
  long i = (long)blockIdx.x * blockDim.x + threadIdx.x;
  long n = (long)Do * Ho * Wo;
  if (i >= n) return;
  int ow = (int)(i % Wo);
  int oh = (int)((i / Wo) % Ho);
  int od = (int)(i / ((long)Wo * Ho));
  float m = 0.f;
  for (int a = 0; a < wd; ++a)
    for (int b2 = 0; b2 < wh; ++b2)
      for (int c = 0; c < ww; ++c) {
        int id = od * sd + a, ih = oh * sh + b2, iw = ow * sw + c;
        if (id < D && ih < H && iw < W)
          m = fmaxf(m, in[((long)id * H + ih) * W + iw]);
      }
  out[i] = m;
}

__global__ void bnrelu_write_kernel(const float* __restrict__ x,
                                    const float* __restrict__ m,
                                    const float* __restrict__ scale,
                                    const float* __restrict__ bias,
                                    float* __restrict__ out, int C, long V) {
  long i = (long)blockIdx.x * blockDim.x + threadIdx.x;
  if (i >= (long)C * V) return;
  long v = i % V;
  int  c = (int)(i / V);
  out[i] = fmaxf(x[i] * scale[c] + bias[c], 0.0f) * m[v];
}

// ---------------------- weight repack (fragment order) ---------------------
// wpak[((tap*nblk+cb)*mtp+mt)*512 + lane*16 + e]; tap>=ntaps or co>=Cout or
// ci>=Cin pack as zero (guard-free padded WMMA tiles).
__global__ void pack_weights_kernel(const float* __restrict__ wgt,
                                    __bf16* __restrict__ wp,
                                    int Cin, int Cout, int ntaps,
                                    int ntp, int mtp) {
  int nblk = (Cin + CBK - 1) / CBK;
  long total = (long)ntp * nblk * mtp * 512;
  long i = (long)blockIdx.x * blockDim.x + threadIdx.x;
  if (i >= total) return;
  int  e    = (int)(i & 15);
  int  lane = (int)((i >> 4) & 31);
  long r    = i >> 9;
  int  mt   = (int)(r % mtp); r /= mtp;
  int  cb   = (int)(r % nblk);
  int  tap  = (int)(r / nblk);
  int half = lane >> 4, m = lane & 15;
  int kl = half * 16 + e;
  int ci = cb * CBK + kl;
  int co = mt * 16 + m;
  float wf = (tap < ntaps && co < Cout && ci < Cin)
                 ? wgt[((long)co * Cin + ci) * ntaps + tap] : 0.0f;
  wp[i] = (__bf16)wf;
}

// ------------------- 3x3x3 stride-1 SAME conv (LDS + WMMA) -----------------

template <int MT>   // M tiles = Cout/16: 1, 2, 4, or 8
__global__ __launch_bounds__(128)
void conv3_wmma_lds_kernel(const float* __restrict__ x, const float* __restrict__ inmask,
                           const float* __restrict__ scale, const float* __restrict__ bias,
                           const __bf16* __restrict__ wpak, const float* __restrict__ resid,
                           const float* __restrict__ omask, float* __restrict__ out,
                           int Cin, int D, int H, int W, int Cout,
                           int use_bn, int use_mask) {
  constexpr int JT = (MT >= 4) ? (MT / 4) : 1;   // accumulators per wave
  __shared__ float  s_raw[ROWS3 * CBP];          // also reused as reduction scratch
  __shared__ __bf16 s_b[ROWS3 * BP];
  __shared__ float  s_msk[ROWS3];
  __shared__ float  s_sc[CBK], s_bi[CBK];

  const int t    = threadIdx.x;
  const int lane = t & 31;
  const int wv   = t >> 5;
  const int half = lane >> 4;
  const int l16  = lane & 15;

  const int od = blockIdx.z, oh = blockIdx.y;
  const int w0 = blockIdx.x * WT;

  const int nblk = (Cin + CBK - 1) / CBK;
  const long HW   = (long)H * W;
  const long DHWl = (long)D * HW;

  // ---- decode this thread's patch rows once (<=2), stage mask rows ----
  int  nrows = 0;
  int  rrow[2]; long rbase[2]; bool rok[2];
  for (int row = t; row < ROWS3; row += 128) {
    int dz = row / 54, rem = row - dz * 54, dy = rem / 18, wx = rem - dy * 18;
    int id = od + dz - 1, ih = oh + dy - 1, iw = w0 + wx - 1;
    bool ok = (id >= 0) && (id < D) && (ih >= 0) && (ih < H) && (iw >= 0) && (iw < W);
    long base = ok ? (id * HW + (long)ih * W + iw) : 0;
    float m = 0.0f;
    if (ok) m = use_mask ? inmask[base] : 1.0f;
    s_msk[row] = m;
    rrow[nrows] = row; rok[nrows] = ok; rbase[nrows] = base;
    ++nrows;
  }

  v8f acc[JT];
#pragma unroll
  for (int j = 0; j < JT; ++j)
#pragma unroll
    for (int r = 0; r < 8; ++r) acc[j][r] = 0.0f;

  const long tapstride = (long)nblk * MT * 512;

  for (int cb = 0; cb < nblk; ++cb) {
    __syncthreads();                     // consumers of previous tile done
    if (t < CBK) {
      int ci = cb * CBK + t;
      s_sc[t] = (use_bn && ci < Cin) ? scale[ci] : 1.0f;
      s_bi[t] = (use_bn && ci < Cin) ? bias[ci]  : 0.0f;
    }
    if (t == 0 && cb + 1 < nblk)        // global_prefetch_b8 hint
      __builtin_prefetch(&wpak[(long)(cb + 1) * MT * 512], 0, 1);

    // ---- stage raw fp32 patch (async-to-LDS when available) ----
    int chmax = Cin - cb * CBK; if (chmax > CBK) chmax = CBK;
    for (int rr = 0; rr < nrows; ++rr) {
      float* dst = &s_raw[rrow[rr] * CBP];
      if (rok[rr]) {
        const float* src = x + (long)cb * CBK * DHWl + rbase[rr];
        int ch = 0;
        for (; ch < chmax; ++ch) {
#if HAVE_ASYNC
          __builtin_amdgcn_global_load_async_to_lds_b32(
              (gptr_i32)const_cast<float*>(src), (lptr_i32)(dst + ch), 0, 0);
#else
          dst[ch] = *src;
#endif
          src += DHWl;
        }
        for (; ch < CBK; ++ch) dst[ch] = 0.0f;
      } else {
#pragma unroll
        for (int ch = 0; ch < CBK; ++ch) dst[ch] = 0.0f;
      }
    }
#if HAVE_ASYNC
    __builtin_amdgcn_s_wait_asynccnt(0);
#endif
    __syncthreads();

    // ---- transform once: BN+ReLU+mask, fp32 -> bf16 tile ----
    for (int rr = 0; rr < nrows; ++rr) {
      int row = rrow[rr];
      float mrow = s_msk[row];
      if (use_bn) {
#pragma unroll
        for (int ch = 0; ch < CBK; ++ch) {
          float v = fmaxf(s_raw[row * CBP + ch] * s_sc[ch] + s_bi[ch], 0.0f) * mrow;
          s_b[row * BP + ch] = (__bf16)v;
        }
      } else {
#pragma unroll
        for (int ch = 0; ch < CBK; ++ch)
          s_b[row * BP + ch] = (__bf16)(s_raw[row * CBP + ch] * mrow);
      }
    }
    __syncthreads();

    // ---- K-loop: compile-time uniform, no guards around WMMA ----
    const __bf16* wcb = wpak + (long)cb * MT * 512;
    if constexpr (MT >= 4) {
      // waves own disjoint M tiles; all 27 taps per wave
#pragma unroll
      for (int tap = 0; tap < 27; ++tap) {
        const int tz = tap / 9, ty = (tap / 3) % 3, tx = tap % 3;
        const int row = (tz * 3 + ty) * 18 + l16 + tx;
        v16bf bf = *(const v16bf*)&s_b[row * BP + half * 16];
#pragma unroll
        for (int j = 0; j < JT; ++j) {
          const int mt = wv + 4 * j;
          v16bf af = *(const v16bf*)&wcb[(long)tap * tapstride + (long)mt * 512 + lane * 16];
          acc[j] = __builtin_amdgcn_wmma_f32_16x16x32_bf16(
              false, af, false, bf, (short)0, acc[j], false, false);
        }
      }
    } else {
      // split-K across waves (28 padded taps; tap 27 has zero weights)
      constexpr int TPW = (MT == 1) ? 7 : 14;
      const int tbase = (MT == 1) ? (wv * 7) : ((wv >> 1) * 14);
      const int mt = (MT == 2) ? (wv & 1) : 0;
#pragma unroll
      for (int k = 0; k < TPW; ++k) {
        int tap = tbase + k;
        int tc = (tap < 27) ? tap : 0;          // B row for pad tap: A is zero
        int tz = tc / 9, ty = (tc / 3) % 3, tx = tc % 3;
        int row = (tz * 3 + ty) * 18 + l16 + tx;
        v16bf bf = *(const v16bf*)&s_b[row * BP + half * 16];
        v16bf af = *(const v16bf*)&wcb[(long)tap * tapstride + (long)mt * 512 + lane * 16];
        acc[0] = __builtin_amdgcn_wmma_f32_16x16x32_bf16(
            false, af, false, bf, (short)0, acc[0], false, false);
      }
    }
  }

  // ---- epilogue ----
  const long oHW = (long)H * W;
  const int  ow  = w0 + l16;
  float om = 1.0f;
  if (omask && ow < W) om = omask[od * oHW + (long)oh * W + ow];

  if constexpr (MT >= 4) {
#pragma unroll
    for (int j = 0; j < JT; ++j) {
      const int mt = wv + 4 * j;
#pragma unroll
      for (int r = 0; r < 8; ++r) {
        int co = mt * 16 + half * 8 + r;
        if (co < Cout && ow < W) {
          long oidx = (long)co * DHWl + od * oHW + (long)oh * W + ow;
          float v = acc[0 + j][r] * om;
          if (resid) v += resid[oidx];
          out[oidx] = v;
        }
      }
    }
  } else {
    // cross-wave K-split reduction through LDS (reuse s_raw)
    float* s_red = s_raw;
    __syncthreads();
#pragma unroll
    for (int r = 0; r < 8; ++r) s_red[wv * 256 + lane * 8 + r] = acc[0][r];
    __syncthreads();
    const int nw = (MT == 1) ? 1 : 2;    // waves that finalize
    if (wv < nw) {
      const int mt = wv;                  // MT==1 -> wv==0 -> mt 0
#pragma unroll
      for (int r = 0; r < 8; ++r) {
        float v;
        if constexpr (MT == 1) {
          v = s_red[0 * 256 + lane * 8 + r] + s_red[1 * 256 + lane * 8 + r] +
              s_red[2 * 256 + lane * 8 + r] + s_red[3 * 256 + lane * 8 + r];
        } else {
          v = s_red[wv * 256 + lane * 8 + r] + s_red[(wv + 2) * 256 + lane * 8 + r];
        }
        int co = mt * 16 + half * 8 + r;
        if (co < Cout && ow < W) {
          long oidx = (long)co * DHWl + od * oHW + (long)oh * W + ow;
          v *= om;
          if (resid) v += resid[oidx];
          out[oidx] = v;
        }
      }
    }
  }
}

// -------------- generic direct conv (strided / (K,1,1) shapes) -------------

template <int JT>   // padded M tiles = 4*JT, zero A-fragments beyond Cout
__global__ __launch_bounds__(128)
void conv_wmma_direct_kernel(const float* __restrict__ x, const float* __restrict__ inmask,
                             const float* __restrict__ scale, const float* __restrict__ bias,
                             const __bf16* __restrict__ wpak, const float* __restrict__ resid,
                             const float* __restrict__ omask, float* __restrict__ out,
                             int Cin, int D, int H, int W,
                             int Cout, int Do, int Ho, int Wo,
                             int kd, int kh, int kw, int stride, int pad) {
  const int t    = threadIdx.x;
  const int lane = t & 31;
  const int wv   = t >> 5;
  const int half = lane >> 4;
  const int l16  = lane & 15;

  const int od = blockIdx.z, oh = blockIdx.y;
  const int w0 = blockIdx.x * WT;

  const int ntaps = kd * kh * kw;
  const int nblk  = (Cin + CBK - 1) / CBK;
  const int mtp   = 4 * JT;

  const long HW   = (long)H * W;
  const long DHWl = (long)D * HW;

  v8f acc[JT];
#pragma unroll
  for (int j = 0; j < JT; ++j)
#pragma unroll
    for (int r = 0; r < 8; ++r) acc[j][r] = 0.0f;

  for (int cb = 0; cb < nblk; ++cb) {
    for (int tap = 0; tap < ntaps; ++tap) {
      int tz = tap / (kh * kw), tr = tap - tz * kh * kw;
      int ty = tr / kw, tx = tr - ty * kw;

      int ow = w0 + l16;
      int id = od * stride - pad + tz;
      int ih = oh * stride - pad + ty;
      int iw = ow * stride - pad + tx;
      bool posok = (ow < Wo) && (id >= 0) && (id < D) && (ih >= 0) && (ih < H) &&
                   (iw >= 0) && (iw < W);
      float mrow = 0.0f;
      if (posok) mrow = inmask ? inmask[id * HW + (long)ih * W + iw] : 1.0f;

      v16bf bf;
#pragma unroll
      for (int e = 0; e < 16; ++e) {
        int kl = half * 16 + e;
        int ci = cb * CBK + kl;
        float val = 0.0f;
        if (posok && ci < Cin) {
          val = x[(long)ci * DHWl + id * HW + (long)ih * W + iw];
          if (scale) val = fmaxf(val * scale[ci] + bias[ci], 0.0f);
          val *= mrow;
        }
        bf[e] = (__bf16)val;
      }

#pragma unroll
      for (int j = 0; j < JT; ++j) {
        int mt = wv + 4 * j;
        v16bf af = *(const v16bf*)&wpak[(((long)tap * nblk + cb) * mtp + mt) * 512 + lane * 16];
        acc[j] = __builtin_amdgcn_wmma_f32_16x16x32_bf16(
            false, af, false, bf, (short)0, acc[j], false, false);
      }
    }
  }

  const long oHW = (long)Ho * Wo;
  int ow = w0 + l16;
  float om = 1.0f;
  if (omask && ow < Wo) om = omask[od * oHW + (long)oh * Wo + ow];
#pragma unroll
  for (int j = 0; j < JT; ++j) {
    int mt = wv + 4 * j;
#pragma unroll
    for (int r = 0; r < 8; ++r) {
      int co = mt * 16 + half * 8 + r;
      if (co < Cout && ow < Wo) {
        long oidx = (long)co * Do * oHW + od * oHW + (long)oh * Wo + ow;
        float v = acc[j][r] * om;
        if (resid) v += resid[oidx];
        out[oidx] = v;
      }
    }
  }
}

// ------------------------------ host orchestration -------------------------

extern "C" void kernel_launch(void* const* d_in, const int* in_sizes, int n_in,
                              void* d_out, int out_size, void* d_ws, size_t ws_size,
                              hipStream_t stream) {
  const float* vf    = (const float*)d_in[0];
  const int*   coors = (const int*)d_in[1];
  // d_in[2] = batch_size (==1)

  // ---- params: JAX pytree leaves, dicts flattened by sorted key ----
  int pi = 3;
  auto nxt = [&]() -> const float* {
    int i = (pi < n_in) ? pi++ : (n_in - 1);
    return (const float*)d_in[i];
  };
  struct BN  { const float *g, *b; };
  auto getBN = [&]() { BN r; r.b = nxt(); r.g = nxt(); return r; };   // 'b' < 'g'
  struct Res { BN bn1, bn2; const float *w1, *w2; };
  auto getRes = [&]() { Res r; r.bn1 = getBN(); r.bn2 = getBN(); r.w1 = nxt(); r.w2 = nxt(); return r; };

  struct FM  { BN a, b2; const float* w; };
  FM fm[4];
  for (int k = 0; k < 3; ++k) { fm[k].a = getBN(); fm[k].b2 = getBN(); fm[k].w = nxt(); }
  fm[3].a = getBN();
  struct Lvl { Res blk[2]; BN pre; const float* wdown; };
  Lvl lv[4];
  for (int k = 0; k < 4; ++k) {
    lv[k].blk[0] = getRes(); lv[k].blk[1] = getRes();
    if (k > 0) { lv[k].pre = getBN(); lv[k].wdown = nxt(); }
  }
  struct Stg { Res blk[2]; BN bnd; const float* wd; };
  Stg st3[3];
  for (int i = 0; i < 3; ++i) {
    st3[i].blk[0] = getRes(); st3[i].blk[1] = getRes();
    st3[i].bnd = getBN(); st3[i].wd = nxt();
  }
  const float* w_in = nxt();

  // ---- workspace layout ----
  float* W0 = (float*)d_ws;
  const size_t ACTCAP = 33000000;               // >= 16*127^3
  float* act[3] = { W0, W0 + ACTCAP, W0 + 2 * ACTCAP };
  float* mAbuf  = W0 + 3 * ACTCAP;              // 127^3 mask
  float* mBbuf  = mAbuf + 2100000;
  float* m2buf  = mBbuf + 2100000;
  float* fbuf   = m2buf + 8192;
  float* stats  = fbuf + 8192;
  float* sscale = stats + 512;
  float* sbias  = sscale + 128;
  __bf16* wpak  = (__bf16*)(sbias + 128);       // <= 28*4*8*512 bf16 (~0.9MB)

  auto zero = [&](float* p, long n) {
    zero_kernel<<<dim3((unsigned)((n + 255) / 256)), 256, 0, stream>>>(p, n);
  };
  auto run_stats = [&](const float* xx, const float* mm, int C, long V) {
    zero(stats, 1 + 2 * (long)C);
    long gb = (V + 255) / 256; if (gb > 256) gb = 256;
    stats_kernel<<<dim3((unsigned)gb, (unsigned)C), 256, 0, stream>>>(xx, mm, V, C, stats);
  };
  auto run_fin = [&](BN bn, int C) {
    finalize_bn_kernel<<<dim3((unsigned)((C + 63) / 64)), 64, 0, stream>>>(
        stats, bn.g, bn.b, sscale, sbias, C);
  };
  auto conv = [&](const float* xx, const float* im, bool bnin, const float* wg,
                  const float* rs, const float* om, float* oo,
                  int Cin, int D, int H, int Wd, int Cout, int Do, int Ho, int Wo,
                  int kd, int kh, int kw, int s2, int pad) {
    int ntaps  = kd * kh * kw;
    int nblk   = (Cin + CBK - 1) / CBK;
    int mtiles = (Cout + 15) / 16;
    bool lds3  = (s2 == 1 && kd == 3 && kh == 3 && kw == 3 && pad == 1);
    int mtp = lds3 ? mtiles : ((mtiles + 3) / 4) * 4;
    int ntp = lds3 ? NTAP3 : ntaps;
    long tot = (long)ntp * nblk * mtp * 512;
    pack_weights_kernel<<<dim3((unsigned)((tot + 255) / 256)), 256, 0, stream>>>(
        wg, wpak, Cin, Cout, ntaps, ntp, mtp);
    dim3 g((unsigned)((Wo + WT - 1) / WT), (unsigned)Ho, (unsigned)Do);
    const float* sc = bnin ? sscale : nullptr;
    const float* bi = bnin ? sbias : nullptr;
    if (lds3) {
      switch (mtiles) {
        case 1:
          conv3_wmma_lds_kernel<1><<<g, 128, 0, stream>>>(xx, im, sc, bi, wpak, rs, om, oo,
              Cin, D, H, Wd, Cout, bnin ? 1 : 0, im ? 1 : 0); break;
        case 2:
          conv3_wmma_lds_kernel<2><<<g, 128, 0, stream>>>(xx, im, sc, bi, wpak, rs, om, oo,
              Cin, D, H, Wd, Cout, bnin ? 1 : 0, im ? 1 : 0); break;
        case 4:
          conv3_wmma_lds_kernel<4><<<g, 128, 0, stream>>>(xx, im, sc, bi, wpak, rs, om, oo,
              Cin, D, H, Wd, Cout, bnin ? 1 : 0, im ? 1 : 0); break;
        default:
          conv3_wmma_lds_kernel<8><<<g, 128, 0, stream>>>(xx, im, sc, bi, wpak, rs, om, oo,
              Cin, D, H, Wd, Cout, bnin ? 1 : 0, im ? 1 : 0); break;
      }
    } else {
      if (mtp <= 4)
        conv_wmma_direct_kernel<1><<<g, 128, 0, stream>>>(xx, im, sc, bi, wpak, rs, om, oo,
            Cin, D, H, Wd, Cout, Do, Ho, Wo, kd, kh, kw, s2, pad);
      else
        conv_wmma_direct_kernel<2><<<g, 128, 0, stream>>>(xx, im, sc, bi, wpak, rs, om, oo,
            Cin, D, H, Wd, Cout, Do, Ho, Wo, kd, kh, kw, s2, pad);
    }
  };

  // ---- forward ----
  const int DHW0 = 127;
  long V0 = (long)DHW0 * DHW0 * DHW0;
  zero(act[0], V0);
  zero(mAbuf, V0);
  scatter_kernel<<<dim3((100000 + 255) / 256), 256, 0, stream>>>(
      vf, coors, act[0], mAbuf, 100000, DHW0, DHW0, DHW0);

  int cur = 0;
  int C = 16, Dd = DHW0;
  float* mcur = mAbuf; float* mnext = mBbuf;

  conv(act[0], nullptr, false, w_in, nullptr, mcur, act[1],
       1, Dd, Dd, Dd, 16, Dd, Dd, Dd, 3, 3, 3, 1, 1);
  cur = 1;

  auto resblock = [&](Res r) {
    long V = (long)Dd * Dd * Dd;
    float* A  = act[cur];
    float* B  = act[(cur + 1) % 3];
    float* Cc = act[(cur + 2) % 3];
    run_stats(A, mcur, C, V); run_fin(r.bn1, C);
    conv(A, mcur, true, r.w1, nullptr, mcur, B,
         C, Dd, Dd, Dd, C, Dd, Dd, Dd, 3, 3, 3, 1, 1);
    run_stats(B, mcur, C, V); run_fin(r.bn2, C);
    conv(B, mcur, true, r.w2, A, mcur, Cc,
         C, Dd, Dd, Dd, C, Dd, Dd, Dd, 3, 3, 3, 1, 1);
    cur = (cur + 2) % 3;
  };
  auto down = [&](BN bn, const float* wd, int Cout) {
    long V = (long)Dd * Dd * Dd;
    int Dn = (Dd - 3) / 2 + 1;
    run_stats(act[cur], mcur, C, V); run_fin(bn, C);
    long Vn = (long)Dn * Dn * Dn;
    pool_mask_kernel<<<dim3((unsigned)((Vn + 255) / 256)), 256, 0, stream>>>(
        mcur, mnext, Dd, Dd, Dd, Dn, Dn, Dn, 3, 3, 3, 2, 2, 2);
    float* B = act[(cur + 1) % 3];
    conv(act[cur], mcur, true, wd, nullptr, mnext, B,
         C, Dd, Dd, Dd, Cout, Dn, Dn, Dn, 3, 3, 3, 2, 0);
    cur = (cur + 1) % 3; C = Cout; Dd = Dn;
    float* tmp = mcur; mcur = mnext; mnext = tmp;
  };

  const int COUTS[3] = { 32, 64, 128 };
  for (int i = 0; i < 3; ++i) {
    resblock(st3[i].blk[0]);
    resblock(st3[i].blk[1]);
    down(st3[i].bnd, st3[i].wd, COUTS[i]);
  }

  float* outp = (float*)d_out;
  long ooff = 0;
  const int KD3[3]  = { 15, 7, 3 };
  const int FMO[4]  = { 32, 64, 96, 128 };
  for (int k = 0; k < 4; ++k) {
    if (k > 0) down(lv[k].pre, lv[k].wdown, 128);
    resblock(lv[k].blk[0]);
    resblock(lv[k].blk[1]);
    long V = (long)Dd * Dd * Dd;
    if (k < 3) {
      run_stats(act[cur], mcur, 128, V); run_fin(fm[k].a, 128);
      int Dz = Dd - KD3[k] + 1;   // == 1
      conv(act[cur], mcur, true, fm[k].w, nullptr, nullptr, fbuf,
           128, Dd, Dd, Dd, FMO[k], Dz, Dd, Dd, KD3[k], 1, 1, 1, 0);
      long V2 = (long)Dz * Dd * Dd;
      pool_mask_kernel<<<dim3((unsigned)((V2 + 255) / 256)), 256, 0, stream>>>(
          mcur, m2buf, Dd, Dd, Dd, Dz, Dd, Dd, KD3[k], 1, 1, 1, 1, 1);
      run_stats(fbuf, m2buf, FMO[k], V2); run_fin(fm[k].b2, FMO[k]);
      long n = (long)FMO[k] * V2;
      bnrelu_write_kernel<<<dim3((unsigned)((n + 255) / 256)), 256, 0, stream>>>(
          fbuf, m2buf, sscale, sbias, outp + ooff, FMO[k], V2);
      ooff += n;
    } else {
      run_stats(act[cur], mcur, 128, V); run_fin(fm[3].a, 128);
      long n = 128 * V;
      bnrelu_write_kernel<<<dim3((unsigned)((n + 255) / 256)), 256, 0, stream>>>(
          act[cur], mcur, sscale, sbias, outp + ooff, 128, V);
      ooff += n;
    }
  }
}
